// LSTMCell_assignment_25950192402731
// MI455X (gfx1250) — matrix operand order repaired and checked by reading, not compiled, not run on hardware
//
#include <hip/hip_runtime.h>
#include <hip/hip_bf16.h>

// ---------------------------------------------------------------------------
// LSTM cell, B=4096, I=H=2048.
// gates[B,4H] = [x|h] @ [Wx|Wh]^T  (single GEMM, M=4096, N=8192, K=4096)
// f32 -> bf16 conversion once into workspace, then a fused
// v_wmma_f32_16x16x32_bf16 GEMM. B tiles staged in LDS via async global->LDS
// (ASYNCcnt), double-buffered with BK=64 stages; B fed to WMMA through a
// 3-deep ds_load_b128 software pipeline; A prefetched one stage ahead.
// Epilogue fuses bias + sigmoid/tanh + cell update.
// ---------------------------------------------------------------------------

typedef __attribute__((ext_vector_type(16))) __bf16 v16bf;
typedef __attribute__((ext_vector_type(8)))  __bf16 v8bf;
typedef __attribute__((ext_vector_type(4)))  __bf16 v4bf;
typedef __attribute__((ext_vector_type(8)))  float  v8f;

// Types matching the async-LDS builtin signature exactly:
typedef int v4i_vs __attribute__((__vector_size__(16)));
typedef __attribute__((address_space(1))) v4i_vs glb_v4i;
typedef __attribute__((address_space(3))) v4i_vs lds_v4i;

#define BSZ 4096
#define HSZ 2048
#define KSZ 4096   // I + H (concatenated K)
#define NG  8192   // 4*H

#define BK        64                  // K elements staged per stage
#define B_PITCH   144                 // bytes per staged 64-elem bf16 row (128B + bank pad)
#define B_BUFSZ   (256 * B_PITCH)     // 36 KB per buffer
#define SMEM_SZ   (2 * B_BUFSZ)       // 72 KB double-buffered

#if __has_builtin(__builtin_amdgcn_global_load_async_to_lds_b128)
#define HAVE_ASYNC_LDS 1
#else
#define HAVE_ASYNC_LDS 0
#endif

// Concatenate two [rows, 2048] f32 matrices along dim1 into [rows, 4096] bf16.
__global__ __launch_bounds__(256) void cvt_cat_kernel(const float* __restrict__ a0,
                                                      const float* __restrict__ a1,
                                                      __bf16* __restrict__ dst,
                                                      int rows) {
  long total4 = (long)rows * KSZ / 4;
  for (long i = (long)blockIdx.x * blockDim.x + threadIdx.x; i < total4;
       i += (long)gridDim.x * blockDim.x) {
    long e = i * 4;            // 4-aligned; 2048 % 4 == 0 so no boundary straddle
    long r = e >> 12;          // / 4096
    long k = e & 4095;
    const float* src = (k < HSZ) ? (a0 + r * HSZ + k) : (a1 + r * HSZ + (k - HSZ));
    float4 v = *(const float4*)src;
    v4bf o;
    o.x = (__bf16)v.x; o.y = (__bf16)v.y; o.z = (__bf16)v.z; o.w = (__bf16)v.w;
    *(v4bf*)(dst + e) = o;
  }
}

__device__ __forceinline__ float sigf(float x) {
  return 1.0f / (1.0f + __expf(-x));
}

__device__ __forceinline__ void wait_async_8() {
#if HAVE_ASYNC_LDS
#if __has_builtin(__builtin_amdgcn_s_wait_asynccnt)
  __builtin_amdgcn_s_wait_asynccnt(8);
#else
  asm volatile("s_wait_asynccnt 0x8" ::: "memory");
#endif
#endif
}

__device__ __forceinline__ void wait_async_0() {
#if HAVE_ASYNC_LDS
#if __has_builtin(__builtin_amdgcn_s_wait_asynccnt)
  __builtin_amdgcn_s_wait_asynccnt(0);
#else
  asm volatile("s_wait_asynccnt 0x0" ::: "memory");
#endif
#endif
}

// Copy one 128-byte B row segment (64 bf16 at gsrc) into LDS at lds_generic.
__device__ __forceinline__ void stage_row(const __bf16* gsrc, char* lds_generic) {
#if HAVE_ASYNC_LDS
  glb_v4i* g = (glb_v4i*)(unsigned long long)(uintptr_t)gsrc;
  lds_v4i* l = (lds_v4i*)(unsigned)(uintptr_t)lds_generic;  // low 32 bits = LDS offset
#pragma unroll
  for (int q = 0; q < 8; ++q)
    __builtin_amdgcn_global_load_async_to_lds_b128(g + q, l + q, 0, 0);
#else
  uint4 q[8];
#pragma unroll
  for (int i = 0; i < 8; ++i) q[i] = *(const uint4*)(gsrc + 8 * i);
#pragma unroll
  for (int i = 0; i < 8; ++i) *(uint4*)(lds_generic + 16 * i) = q[i];
#endif
}

// Assemble a 32x16 B operand from LDS (two 16B halves 32B apart per lane).
__device__ __forceinline__ v16bf load_btile(const char* p) {
  v8bf lo = *(const v8bf*)p;
  v8bf hi = *(const v8bf*)(p + 32);
  return __builtin_shufflevector(lo, hi,
      0,1,2,3,4,5,6,7,8,9,10,11,12,13,14,15);
}

__device__ __forceinline__ v16bf cat_a(v8bf lo, v8bf hi) {
  return __builtin_shufflevector(lo, hi,
      0,1,2,3,4,5,6,7,8,9,10,11,12,13,14,15);
}

// Grid: (4096/128, 2048/64). Block: 256 threads = 8 waves.
// Wave w owns rows [m0, m0+16); block owns cols [j0, j0+64) of each of 4 gates.
__global__ __launch_bounds__(256) void lstm_wmma_kernel(
    const __bf16* __restrict__ A,    // [4096][4096] bf16 ([x|h])
    const __bf16* __restrict__ W,    // [8192][4096] bf16 ([Wx|Wh], gate-major rows)
    const float*  __restrict__ c_in, // [4096][2048]
    const float*  __restrict__ bx,   // [8192]
    const float*  __restrict__ bh,   // [8192]
    float* __restrict__ h_out,       // [4096][2048]
    float* __restrict__ c_out)       // [4096][2048]
{
  __shared__ __attribute__((aligned(16))) char smem[SMEM_SZ];

  const int tid     = threadIdx.x;
  const int lane    = tid & 31;
  const int wave    = tid >> 5;
  const int m0      = blockIdx.x * 128 + wave * 16;
  const int j0      = blockIdx.y * 64;
  const int halfsel = lane >> 4;   // lane half selects K phase
  const int lrow    = lane & 15;

  // A 16x32 bf16 operand layout: lane (lrow, half) holds row m0+lrow;
  // V0..V3 = K[8*half .. +8), V4..V7 = K[8*half+16 .. +8).
  const __bf16* aptr = A + (size_t)(m0 + lrow) * KSZ + 8 * halfsel;

  // Staging: thread tid owns staged B row tid = g*64 + (t*16 + lrow);
  // global source row = g*2048 + j0 + (tid & 63); 128B (BK=64) per stage.
  const __bf16* gB = W + (size_t)((tid >> 6) * HSZ + j0 + (tid & 63)) * KSZ;
  char* ldsW = smem + tid * B_PITCH;

  // Compute-side LDS read base: row (tile*16 + lrow), byte 16*halfsel within
  // the 32-element substep, substep at +64 bytes.
  const char* ldsR = smem + lrow * B_PITCH + 16 * halfsel;

  v8f acc[4][4];
  v8f zero = {};
#pragma unroll
  for (int g = 0; g < 4; ++g)
#pragma unroll
    for (int t = 0; t < 4; ++t)
      acc[g][t] = zero;

  // Prologue: stage ks=0 into buffer 0; prefetch A for stage 0.
  stage_row(gB, ldsW);
  v8bf a0lo = *(const v8bf*)(aptr + 0);
  v8bf a0hi = *(const v8bf*)(aptr + 16);
  v8bf a1lo = *(const v8bf*)(aptr + 32);
  v8bf a1hi = *(const v8bf*)(aptr + 48);

  for (int ks = 0; ks < KSZ; ks += BK) {
    const int buf = (ks >> 6) & 1;

    v8bf n0lo, n0hi, n1lo, n1hi;
    if (ks + BK < KSZ) {
      // Prefetch next stage: B via async->LDS, A via global loads (consumed
      // one full stage later, so L2 latency is off the critical path).
      stage_row(gB + (ks + BK), ldsW + (buf ^ 1) * B_BUFSZ);
      n0lo = *(const v8bf*)(aptr + ks + BK);
      n0hi = *(const v8bf*)(aptr + ks + BK + 16);
      n1lo = *(const v8bf*)(aptr + ks + BK + 32);
      n1hi = *(const v8bf*)(aptr + ks + BK + 48);
      wait_async_8();   // current buffer's 8 transfers complete (per-wave FIFO)
    } else {
      wait_async_0();
    }
    __syncthreads();    // all waves' staging for `buf` visible

    const v16bf a0 = cat_a(a0lo, a0hi);
    const v16bf a1 = cat_a(a1lo, a1hi);
    const char* lb = ldsR + buf * B_BUFSZ;

    // 32 tiles: linear i -> substep i>>4, tile (i&15) = g*4+t.
    // 3-deep rotating B pipeline keeps two ds_load_b128 pairs in flight
    // ahead of each WMMA (wait dscnt<=4 instead of <=2).
    v16bf bq[3];
    bq[0] = load_btile(lb + 0 * 16 * B_PITCH);
    bq[1] = load_btile(lb + 1 * 16 * B_PITCH);
    bq[2] = load_btile(lb + 2 * 16 * B_PITCH);
#pragma unroll
    for (int i = 0; i < 32; ++i) {
      const int sub = i >> 4;
      const int ii  = i & 15;
      const int slot = i % 3;
      v16bf b = bq[slot];
      if (i + 3 < 32) {
        const int j = i + 3;
        bq[slot] = load_btile(lb + (j & 15) * 16 * B_PITCH + (j >> 4) * 64);
      }
      acc[ii >> 2][ii & 3] = __builtin_amdgcn_wmma_f32_16x16x32_bf16(
          /*neg_a=*/false, sub ? a1 : a0, /*neg_b=*/false, b,
          /*c_mod=*/(short)0, acc[ii >> 2][ii & 3],
          /*reuse_a=*/false, /*reuse_b=*/false);
    }

    a0lo = n0lo; a0hi = n0hi; a1lo = n1lo; a1hi = n1hi;
    __syncthreads();    // everyone done reading `buf` before it is re-staged
  }

  // Epilogue. C/D layout: VGPR r, lanes 0-15 -> (M=r, N=lane),
  // lanes 16-31 -> (M=r+8, N=lane-16).
#pragma unroll
  for (int t = 0; t < 4; ++t) {
    const int j  = j0 + t * 16 + lrow;
    const float bi = bx[0 * HSZ + j] + bh[0 * HSZ + j];
    const float bf = bx[1 * HSZ + j] + bh[1 * HSZ + j];
    const float bg = bx[2 * HSZ + j] + bh[2 * HSZ + j];
    const float bo = bx[3 * HSZ + j] + bh[3 * HSZ + j];
#pragma unroll
    for (int r = 0; r < 8; ++r) {
      const int row = m0 + r + 8 * halfsel;
      const size_t o = (size_t)row * HSZ + j;
      const float iv = sigf(acc[0][t][r] + bi);
      const float fv = sigf(acc[1][t][r] + bf);
      const float gv = tanhf(acc[2][t][r] + bg);
      const float ov = sigf(acc[3][t][r] + bo);
      const float cn = fv * c_in[o] + iv * gv;
      const float hn = ov * tanhf(cn);
      h_out[o] = hn;
      c_out[o] = cn;
    }
  }
}

extern "C" void kernel_launch(void* const* d_in, const int* in_sizes, int n_in,
                              void* d_out, int out_size, void* d_ws, size_t ws_size,
                              hipStream_t stream) {
  const float* x  = (const float*)d_in[0];
  const float* h  = (const float*)d_in[1];
  const float* c  = (const float*)d_in[2];
  const float* Wx = (const float*)d_in[3];
  const float* bx = (const float*)d_in[4];
  const float* Wh = (const float*)d_in[5];
  const float* bh = (const float*)d_in[6];

  float* h_out = (float*)d_out;
  float* c_out = h_out + (size_t)BSZ * HSZ;

  // Workspace: Abf [4096,4096] bf16 (32MB) then Wbf [8192,4096] bf16 (64MB).
  __bf16* Abf = (__bf16*)d_ws;
  __bf16* Wbf = (__bf16*)((char*)d_ws + (size_t)BSZ * KSZ * sizeof(__bf16));

  cvt_cat_kernel<<<2048, 256, 0, stream>>>(x,  h,  Abf, BSZ);
  cvt_cat_kernel<<<4096, 256, 0, stream>>>(Wx, Wh, Wbf, NG);

  dim3 grid(BSZ / 128, HSZ / 64);
  lstm_wmma_kernel<<<grid, 256, 0, stream>>>(Abf, Wbf, c, bx, bh, h_out, c_out);
}
